// LinearAttention_80942953661157
// MI455X (gfx1250) — matrix operand order repaired and checked by reading, not compiled, not run on hardware
//
#include <hip/hip_runtime.h>
#include <hip/hip_bf16.h>

typedef _Float16 f16;
typedef _Float16 v8h  __attribute__((ext_vector_type(8)));
typedef _Float16 v16h __attribute__((ext_vector_type(16)));
typedef float    v8f  __attribute__((ext_vector_type(8)));

#define BATCH  8
#define CH     512
#define SEQ    4096
#define HEADS  8
#define DHEAD  64
#define HID    512
#define OQKV   1536          // q(512) | k(512) | v(512) stacked
#define QSCALE 0.125f        // 64^-0.5
#define SQRT_C 22.627416997969522f  // sqrt(512)

union Frag { v16h v; v8h h[2]; };

// ---------------- 1. transpose + convert x[b][c][n] f32 -> xt[b][n][c] f16
__global__ void k_transpose_x(const float* __restrict__ x, f16* __restrict__ xt) {
  __shared__ f16 tile[32][33];
  int b  = blockIdx.z;
  int c0 = blockIdx.y * 32;
  int n0 = blockIdx.x * 32;
  int tx = threadIdx.x;   // 0..31
  int ty = threadIdx.y;   // 0..7
  const float* xb = x + (size_t)b * CH * SEQ;
  f16* xtb = xt + (size_t)b * SEQ * CH;
#pragma unroll
  for (int j = 0; j < 4; ++j) {
    int r = ty + j * 8;
    tile[r][tx] = (f16)xb[(size_t)(c0 + r) * SEQ + n0 + tx];
  }
  __syncthreads();
#pragma unroll
  for (int j = 0; j < 4; ++j) {
    int r = ty + j * 8;
    xtb[(size_t)(n0 + r) * CH + c0 + tx] = tile[tx][r];
  }
}

// ---------------- 2. pack Wq|Wk|Wv -> f16 [1536][512]
__global__ void k_pack_w(const float* __restrict__ Wq, const float* __restrict__ Wk,
                         const float* __restrict__ Wv, f16* __restrict__ Wc) {
  int i = blockIdx.x * 256 + threadIdx.x;   // 0..786431
  int o = i >> 9;
  float v = (o < 512) ? Wq[i] : (o < 1024) ? Wk[i - 512 * 512] : Wv[i - 1024 * 512];
  Wc[i] = (f16)v;
}

// ---------------- 3. QKV GEMM: qkv[b][o][n] = Wc[o][:] . x[:, n]
// one wave per 64(o) x 16(n) block; unroll-2 pipelined k-loop, hoisted base ptrs
__global__ void k_qkv_gemm(const f16* __restrict__ Wc, const f16* __restrict__ xt,
                           f16* __restrict__ qkv) {
  int lane = threadIdx.x;               // blockDim = 32
  int row  = lane & 15;
  int hi   = lane >> 4;
  int n0 = blockIdx.x * 16;
  int o0 = blockIdx.y * 64;
  int b  = blockIdx.z;
  // B fragment base: lane = column n0+row, K runs at hi*8 and hi*8+16
  const f16* xbase = xt + (size_t)b * SEQ * CH + (size_t)(n0 + row) * CH + hi * 8;
  // A fragment bases: lane = row o0+t*16+row, K runs at hi*8 and hi*8+16
  const f16* abase[4];
#pragma unroll
  for (int t = 0; t < 4; ++t)
    abase[t] = Wc + (size_t)(o0 + t * 16 + row) * CH + hi * 8;

  v8f acc[4] = {};
#pragma unroll 2
  for (int k0 = 0; k0 < CH; k0 += 32) {
    Frag bf;
    bf.h[0] = *(const v8h*)(xbase + k0);
    bf.h[1] = *(const v8h*)(xbase + k0 + 16);
    Frag af[4];
#pragma unroll
    for (int t = 0; t < 4; ++t) {
      af[t].h[0] = *(const v8h*)(abase[t] + k0);
      af[t].h[1] = *(const v8h*)(abase[t] + k0 + 16);
    }
#pragma unroll
    for (int t = 0; t < 4; ++t)
      acc[t] = __builtin_amdgcn_wmma_f32_16x16x32_f16(false, af[t].v, false, bf.v,
                                                      (short)0, acc[t], false, false);
  }
  int mrow = 8 * hi;
  f16* outb = qkv + (size_t)b * OQKV * SEQ;
#pragma unroll
  for (int t = 0; t < 4; ++t)
#pragma unroll
    for (int j = 0; j < 8; ++j)
      outb[(size_t)(o0 + t * 16 + mrow + j) * SEQ + n0 + row] = (f16)acc[t][j];
}

// ---------------- 4. q softmax over d (64), write qt[b][n][h*64+d] * QSCALE
__global__ void k_softmax_q(const f16* __restrict__ qkv, f16* __restrict__ qt) {
  int idx = blockIdx.x * 256 + threadIdx.x;   // b*h*n = 2^18
  int n = idx & (SEQ - 1);
  int h = (idx >> 12) & 7;
  int b = idx >> 15;
  const f16* qcol = qkv + (size_t)b * OQKV * SEQ + (size_t)(h * DHEAD) * SEQ + n;
  float vals[64];
  float m = -1e30f;
#pragma unroll
  for (int d = 0; d < 64; ++d) { vals[d] = (float)qcol[(size_t)d * SEQ]; m = fmaxf(m, vals[d]); }
  float s = 0.f;
#pragma unroll
  for (int d = 0; d < 64; ++d) { vals[d] = __expf(vals[d] - m); s += vals[d]; }
  float inv = QSCALE / s;
  f16* orow = qt + (size_t)b * SEQ * HID + (size_t)n * HID + h * DHEAD;
#pragma unroll
  for (int d = 0; d < 64; ++d) orow[d] = (f16)(vals[d] * inv);
}

// ---------------- 5. k softmax over n (4096), in place on qkv rows [512,1024)
__global__ void k_softmax_k(f16* __restrict__ qkv) {
  int bid = blockIdx.x;        // 0..4095
  int b = bid >> 9;
  int r = bid & 511;
  f16* row = qkv + (size_t)b * OQKV * SEQ + (size_t)(512 + r) * SEQ;
  int t = threadIdx.x;         // 256
  __shared__ float red[256];
  float vals[16];
  float m = -1e30f;
#pragma unroll
  for (int i = 0; i < 16; ++i) { vals[i] = (float)row[t + i * 256]; m = fmaxf(m, vals[i]); }
  red[t] = m; __syncthreads();
  for (int s = 128; s > 0; s >>= 1) { if (t < s) red[t] = fmaxf(red[t], red[t + s]); __syncthreads(); }
  m = red[0]; __syncthreads();
  float sum = 0.f;
#pragma unroll
  for (int i = 0; i < 16; ++i) { vals[i] = __expf(vals[i] - m); sum += vals[i]; }
  red[t] = sum; __syncthreads();
  for (int s = 128; s > 0; s >>= 1) { if (t < s) red[t] += red[t + s]; __syncthreads(); }
  float inv = 1.f / red[0];
#pragma unroll
  for (int i = 0; i < 16; ++i) row[t + i * 256] = (f16)(vals[i] * inv);
}

// ---------------- 6. ctx[b][h][d][e] = sum_n k[d][n] * v[e][n]   (K=4096 WMMA)
__global__ void k_ctx_gemm(const f16* __restrict__ qkv, float* __restrict__ ctx) {
  int bid = blockIdx.x;        // 64 = b*h
  int b = bid >> 3, h = bid & 7;
  int w = threadIdx.x >> 5;    // 16 waves: 4x4 tile grid
  int lane = threadIdx.x & 31;
  int row  = lane & 15;
  int hi   = lane >> 4;
  int di = w >> 2, ei = w & 3;
  const f16* kb = qkv + (size_t)b * OQKV * SEQ + (size_t)(512 + h * DHEAD) * SEQ;
  const f16* vb = qkv + (size_t)b * OQKV * SEQ + (size_t)(1024 + h * DHEAD) * SEQ;
  const f16* abase = kb + (size_t)(di * 16 + row) * SEQ + hi * 8;   // A: k rows
  const f16* bbase = vb + (size_t)(ei * 16 + row) * SEQ + hi * 8;   // B: v rows (= cols)
  v8f acc = {};
#pragma unroll 8
  for (int k0 = 0; k0 < SEQ; k0 += 32) {
    Frag af, bf;
    af.h[0] = *(const v8h*)(abase + k0);
    af.h[1] = *(const v8h*)(abase + k0 + 16);
    bf.h[0] = *(const v8h*)(bbase + k0);
    bf.h[1] = *(const v8h*)(bbase + k0 + 16);
    acc = __builtin_amdgcn_wmma_f32_16x16x32_f16(false, af.v, false, bf.v,
                                                 (short)0, acc, false, false);
  }
  float* cb = ctx + (size_t)(b * 8 + h) * 64 * 64;
  int mrow = di * 16 + 8 * hi;
  int ncol = ei * 16 + row;
#pragma unroll
  for (int j = 0; j < 8; ++j) cb[(size_t)(mrow + j) * 64 + ncol] = acc[j];
}

// ---------------- 7. Mcat[b][o][h*64+d] = sum_e Wout[o][h*64+e] * ctx[b][h][d][e]
__global__ void k_mcat(const float* __restrict__ Wout, const float* __restrict__ ctx,
                       f16* __restrict__ Mc) {
  int idx = blockIdx.x * 256 + threadIdx.x;   // b*512*512 = 2^21
  int hd = idx & 511;
  int o  = (idx >> 9) & 511;
  int b  = idx >> 18;
  int h = hd >> 6, d = hd & 63;
  const float* wrow = Wout + (size_t)o * HID + h * 64;
  const float* crow = ctx + ((size_t)(b * 8 + h) * 64 + d) * 64;
  float s = 0.f;
#pragma unroll
  for (int e = 0; e < 64; ++e) s += wrow[e] * crow[e];
  Mc[idx] = (f16)s;
}

// ---------------- 8. y[b][o][n] = Mcat[b] @ qt[b] + bout   (512x512x4096 WMMA)
__global__ void k_out_gemm(const f16* __restrict__ Mc, const f16* __restrict__ qt,
                           const float* __restrict__ bout, float* __restrict__ y) {
  int lane = threadIdx.x;               // blockDim = 32
  int row  = lane & 15;
  int hi   = lane >> 4;
  int n0 = blockIdx.x * 16;
  int o0 = blockIdx.y * 64;
  int b  = blockIdx.z;
  const f16* qbase = qt + (size_t)b * SEQ * HID + (size_t)(n0 + row) * HID + hi * 8;
  const f16* Mb    = Mc + (size_t)b * HID * HID;
  const f16* abase[4];
#pragma unroll
  for (int t = 0; t < 4; ++t)
    abase[t] = Mb + (size_t)(o0 + t * 16 + row) * HID + hi * 8;

  v8f acc[4] = {};
#pragma unroll 2
  for (int k0 = 0; k0 < HID; k0 += 32) {
    Frag bf;
    bf.h[0] = *(const v8h*)(qbase + k0);
    bf.h[1] = *(const v8h*)(qbase + k0 + 16);
    Frag af[4];
#pragma unroll
    for (int t = 0; t < 4; ++t) {
      af[t].h[0] = *(const v8h*)(abase[t] + k0);
      af[t].h[1] = *(const v8h*)(abase[t] + k0 + 16);
    }
#pragma unroll
    for (int t = 0; t < 4; ++t)
      acc[t] = __builtin_amdgcn_wmma_f32_16x16x32_f16(false, af[t].v, false, bf.v,
                                                      (short)0, acc[t], false, false);
  }
  int mrow = 8 * hi;
  float* yb = y + (size_t)b * CH * SEQ;
#pragma unroll
  for (int t = 0; t < 4; ++t) {
#pragma unroll
    for (int j = 0; j < 8; ++j) {
      int o = o0 + t * 16 + mrow + j;
      yb[(size_t)o * SEQ + n0 + row] = acc[t][j] + bout[o];
    }
  }
}

// ---------------- 9. RMSNorm over channel dim, in place on d_out
__global__ void k_rmsnorm(float* __restrict__ y, const float* __restrict__ g) {
  int idx = blockIdx.x * 256 + threadIdx.x;   // b*n = 32768
  int n = idx & (SEQ - 1);
  int b = idx >> 12;
  float* col = y + (size_t)b * CH * SEQ + n;
  float ss = 0.f;
  for (int c = 0; c < CH; ++c) { float v = col[(size_t)c * SEQ]; ss += v * v; }
  float scale = SQRT_C / fmaxf(sqrtf(ss), 1e-12f);
  for (int c = 0; c < CH; ++c) col[(size_t)c * SEQ] = col[(size_t)c * SEQ] * scale * g[c];
}

extern "C" void kernel_launch(void* const* d_in, const int* in_sizes, int n_in,
                              void* d_out, int out_size, void* d_ws, size_t ws_size,
                              hipStream_t stream) {
  const float* x    = (const float*)d_in[0];
  const float* Wq   = (const float*)d_in[1];
  const float* Wk   = (const float*)d_in[2];
  const float* Wv   = (const float*)d_in[3];
  const float* Wout = (const float*)d_in[4];
  const float* bout = (const float*)d_in[5];
  const float* g    = (const float*)d_in[6];
  float* out = (float*)d_out;

  char* ws = (char*)d_ws;
  f16*   xt  = (f16*)(ws);                        // 32 MB  [b][n][c] f16
  f16*   qkv = (f16*)(ws + 33554432);             // 96 MB  [b][1536][n] f16
  f16*   qt  = (f16*)(ws + 134217728);            // 32 MB  [b][n][512] f16
  float* ctx = (float*)(ws + 167772160);          // 1 MB   [b][h][64][64] f32
  f16*   Mc  = (f16*)(ws + 168820736);            // 4 MB   [b][512][512] f16
  f16*   Wc  = (f16*)(ws + 173015040);            // 1.5 MB [1536][512] f16

  k_transpose_x<<<dim3(SEQ / 32, CH / 32, BATCH), dim3(32, 8), 0, stream>>>(x, xt);
  k_pack_w<<<(OQKV * CH) / 256, 256, 0, stream>>>(Wq, Wk, Wv, Wc);
  k_qkv_gemm<<<dim3(SEQ / 16, OQKV / 64, BATCH), 32, 0, stream>>>(Wc, xt, qkv);
  k_softmax_q<<<(BATCH * HEADS * SEQ) / 256, 256, 0, stream>>>(qkv, qt);
  k_softmax_k<<<BATCH * HID, 256, 0, stream>>>(qkv);
  k_ctx_gemm<<<BATCH * HEADS, 512, 0, stream>>>(qkv, ctx);
  k_mcat<<<(BATCH * CH * HID) / 256, 256, 0, stream>>>(Wout, ctx, Mc);
  k_out_gemm<<<dim3(SEQ / 16, CH / 64, BATCH), 32, 0, stream>>>(Mc, qt, bout, out);
  k_rmsnorm<<<(BATCH * SEQ) / 256, 256, 0, stream>>>(out, g);
}
